// MbpGINELayer_53833120088741
// MI455X (gfx1250) — matrix operand order repaired and checked by reading, not compiled, not run on hardware
//
#include <hip/hip_runtime.h>
#include <hip/hip_bf16.h>

typedef __attribute__((ext_vector_type(16))) _Float16 v16h;
typedef __attribute__((ext_vector_type(8)))  float    v8f;

#define N_NODES 50000
#define N_EDGES 800000
#define H 96

// ---------------------------------------------------------------------------
// D = A(16x32 f16) * B(32x16 f16) + C(f32)   ->  v_wmma_f32_16x16x32_f16
// ---------------------------------------------------------------------------
__device__ __forceinline__ v8f wmma16(v16h a, v16h b, v8f c) {
  return __builtin_amdgcn_wmma_f32_16x16x32_f16(
      /*neg_a=*/false, a, /*neg_b=*/false, b,
      /*c_mod=*/(short)0, c, /*reuse_a=*/false, /*reuse_b=*/false);
}

// A-tile loader (16x32, f16) from row-major f32 source (row stride ld elems),
// column offset ko.  Wave32 A layout: lane L -> row m = L&15;
// halves 0..7  = K = (L>>4)*8 + i ;  halves 8..15 = K = 16 + (L>>4)*8 + i.
__device__ __forceinline__ v16h load_a_f32(const float* __restrict__ src,
                                           int ld, int ko, int lane) {
  const int m  = lane & 15;
  const int kb = (lane >> 4) * 8;
  const float* p = src + (size_t)m * ld + ko + kb;
  v16h a;
#pragma unroll
  for (int i = 0; i < 8; ++i) a[i] = (_Float16)p[i];
#pragma unroll
  for (int i = 0; i < 8; ++i) a[8 + i] = (_Float16)p[16 + i];
  return a;
}

// B-tile loader for C = A * W^T :  B[k][n] = W16[nbase+n][ko+k].
// Wave32 B layout: lane L -> col n = L&15;
// halves 0..7 = K = (L>>4)*8 + i ;  halves 8..15 = K = 16 + (L>>4)*8 + i.
__device__ __forceinline__ v16h load_b_w(const _Float16* __restrict__ W,
                                         int ld, int nbase, int ko, int lane) {
  const int n  = lane & 15;
  const int kb = (lane >> 4) * 8;
  const _Float16* p = W + (size_t)(nbase + n) * ld + ko + kb;
  v16h b;
#pragma unroll
  for (int i = 0; i < 8; ++i) { b[i] = p[i]; b[8 + i] = p[16 + i]; }
  return b;
}

// ---------------------------------------------------------------------------
// prep kernels
// ---------------------------------------------------------------------------
__global__ void zero_f32_kernel(float* __restrict__ p, int n) {
  int i = blockIdx.x * blockDim.x + threadIdx.x;
  if (i < n) p[i] = 0.0f;
}

__global__ void cvt_f16_kernel(const float* __restrict__ s,
                               _Float16* __restrict__ d, int n) {
  int i = blockIdx.x * blockDim.x + threadIdx.x;
  if (i < n) d[i] = (_Float16)s[i];
}

// ---------------------------------------------------------------------------
// QKV = x @ qkv_w^T + qkv_b      (50000x96) x (96x288)
// One wave per 16-row node tile; A reused across 18 output tiles.
// ---------------------------------------------------------------------------
__global__ __launch_bounds__(32) void qkv_gemm_kernel(
    const float* __restrict__ x, const _Float16* __restrict__ w16,
    const float* __restrict__ bias, float* __restrict__ out) {
  const int tile = blockIdx.x;
  const int lane = threadIdx.x;
  const float* xrow = x + (size_t)tile * 16 * H;
  const v16h a0 = load_a_f32(xrow, H, 0, lane);
  const v16h a1 = load_a_f32(xrow, H, 32, lane);
  const v16h a2 = load_a_f32(xrow, H, 64, lane);
  const int col0  = lane & 15;
  const int rbase = (lane >> 4) * 8;
  for (int nt = 0; nt < 18; ++nt) {
    v8f acc = {};
    acc = wmma16(a0, load_b_w(w16, H, nt * 16, 0, lane), acc);
    acc = wmma16(a1, load_b_w(w16, H, nt * 16, 32, lane), acc);
    acc = wmma16(a2, load_b_w(w16, H, nt * 16, 64, lane), acc);
    const int col = nt * 16 + col0;
    const float b = bias[col];
#pragma unroll
    for (int r = 0; r < 8; ++r) {
      const int m = tile * 16 + rbase + r;
      out[(size_t)m * (3 * H) + col] = acc[r] + b;
    }
  }
}

// ---------------------------------------------------------------------------
// Fused edge kernel: one wave per 16 edges.
//  - Ew/Eb = conn @ e_w^T + e_b via WMMA (never materialized in HBM)
//  - gather Q[dst], K[src], V[src]  (coalesced: one row per half-wave)
//  - conn_out = relu(signsqrt((Q+K)*Ew) + Eb)  on D-registers
//  - scatter-add conn_out -> eagg, V[src] -> agg (f32 L2 atomics)
//  - e_out = layernorm(conn + conn_out)  (row stats via shfl_xor in half-wave)
// ---------------------------------------------------------------------------
__global__ __launch_bounds__(32) void edge_kernel(
    const float* __restrict__ conn, const int* __restrict__ edge_index,
    const float* __restrict__ qkv, const _Float16* __restrict__ ew16,
    const float* __restrict__ e_b, const float* __restrict__ ln_g,
    const float* __restrict__ ln_b, float* __restrict__ eagg,
    float* __restrict__ agg, float* __restrict__ e_out) {
  __shared__ float lds_e[16 * H];
  __shared__ float lds_mu[16];
  __shared__ float lds_rs[16];
  __shared__ int sh_dst[16];
  __shared__ int sh_src[16];

  const int tile = blockIdx.x;
  const int lane = threadIdx.x;
  const int e0 = tile * 16;

  if (lane < 16) sh_dst[lane] = edge_index[e0 + lane];
  else           sh_src[lane - 16] = edge_index[N_EDGES + e0 + (lane - 16)];
  __syncthreads();

  const float* crow = conn + (size_t)e0 * H;
  const v16h a0 = load_a_f32(crow, H, 0, lane);
  const v16h a1 = load_a_f32(crow, H, 32, lane);
  const v16h a2 = load_a_f32(crow, H, 64, lane);

  const int hh = lane >> 4;
  const int col0 = lane & 15;
  float rowsum[8], rowsq[8];
#pragma unroll
  for (int r = 0; r < 8; ++r) { rowsum[r] = 0.0f; rowsq[r] = 0.0f; }

  for (int nt = 0; nt < 6; ++nt) {
    v8f accw = {}, accb = {};
    accw = wmma16(a0, load_b_w(ew16, H, nt * 16, 0, lane), accw);
    accw = wmma16(a1, load_b_w(ew16, H, nt * 16, 32, lane), accw);
    accw = wmma16(a2, load_b_w(ew16, H, nt * 16, 64, lane), accw);
    accb = wmma16(a0, load_b_w(ew16, H, H + nt * 16, 0, lane), accb);
    accb = wmma16(a1, load_b_w(ew16, H, H + nt * 16, 32, lane), accb);
    accb = wmma16(a2, load_b_w(ew16, H, H + nt * 16, 64, lane), accb);
    const int j = nt * 16 + col0;
    const float bw = e_b[j];
    const float bb = e_b[H + j];
#pragma unroll
    for (int r = 0; r < 8; ++r) {
      const int i = hh * 8 + r;
      const int d = sh_dst[i];
      const int s = sh_src[i];
      const float qv = qkv[(size_t)d * (3 * H) + j];
      const float kv = qkv[(size_t)s * (3 * H) + H + j];
      const float vv = qkv[(size_t)s * (3 * H) + 2 * H + j];
      const float ew = accw[r] + bw;
      const float eb = accb[r] + bb;
      const float c1 = (qv + kv) * ew;
      const float c2 = copysignf(sqrtf(fabsf(c1)), c1);
      const float co = fmaxf(c2 + eb, 0.0f);
      const float ev = crow[(size_t)i * H + j] + co;
      lds_e[i * H + j] = ev;
      rowsum[r] += ev;
      rowsq[r]  += ev * ev;
      atomicAdd(&eagg[(size_t)d * H + j], co);
      atomicAdd(&agg[(size_t)d * H + j], vv);
    }
  }

#pragma unroll
  for (int r = 0; r < 8; ++r) {
    float s = rowsum[r], q = rowsq[r];
#pragma unroll
    for (int m = 1; m < 16; m <<= 1) {
      s += __shfl_xor(s, m, 32);
      q += __shfl_xor(q, m, 32);
    }
    if (col0 == 0) {
      const int i = hh * 8 + r;
      const float mu = s * (1.0f / H);
      const float var = q * (1.0f / H) - mu * mu;
      lds_mu[i] = mu;
      lds_rs[i] = rsqrtf(var + 1e-5f);
    }
  }
  __syncthreads();
  for (int t = lane; t < 16 * H; t += 32) {
    const int i = t / H;
    const int j = t - i * H;
    e_out[(size_t)e0 * H + t] =
        (lds_e[t] - lds_mu[i]) * lds_rs[i] * ln_g[j] + ln_b[j];
  }
}

// ---------------------------------------------------------------------------
// Node kernel: h = layernorm(x + (agg + (eagg @ cl_w^T + cl_b)) @ nl_w^T + nl_b)
// Two chained WMMA GEMMs per 16-node tile; intermediate routed via LDS.
// ---------------------------------------------------------------------------
__global__ __launch_bounds__(32) void node_kernel(
    const float* __restrict__ x, const float* __restrict__ agg,
    const float* __restrict__ eagg, const _Float16* __restrict__ clw16,
    const float* __restrict__ clb, const _Float16* __restrict__ nlw16,
    const float* __restrict__ nlb, const float* __restrict__ ln_g,
    const float* __restrict__ ln_b, float* __restrict__ h_out) {
  __shared__ float lds_t[16 * H];
  __shared__ float lds_h[16 * H];
  __shared__ float lds_mu[16];
  __shared__ float lds_rs[16];

  const int tile = blockIdx.x;
  const int lane = threadIdx.x;
  const int hh = lane >> 4;
  const int col0 = lane & 15;
  const size_t row0 = (size_t)tile * 16;

  const float* erow = eagg + row0 * H;
  v16h a0 = load_a_f32(erow, H, 0, lane);
  v16h a1 = load_a_f32(erow, H, 32, lane);
  v16h a2 = load_a_f32(erow, H, 64, lane);

  for (int nt = 0; nt < 6; ++nt) {
    v8f acc = {};
    acc = wmma16(a0, load_b_w(clw16, H, nt * 16, 0, lane), acc);
    acc = wmma16(a1, load_b_w(clw16, H, nt * 16, 32, lane), acc);
    acc = wmma16(a2, load_b_w(clw16, H, nt * 16, 64, lane), acc);
    const int j = nt * 16 + col0;
    const float b = clb[j];
#pragma unroll
    for (int r = 0; r < 8; ++r) {
      const int i = hh * 8 + r;
      lds_t[i * H + j] = acc[r] + b + agg[(row0 + i) * H + j];
    }
  }
  __syncthreads();

  a0 = load_a_f32(lds_t, H, 0, lane);
  a1 = load_a_f32(lds_t, H, 32, lane);
  a2 = load_a_f32(lds_t, H, 64, lane);

  float rowsum[8], rowsq[8];
#pragma unroll
  for (int r = 0; r < 8; ++r) { rowsum[r] = 0.0f; rowsq[r] = 0.0f; }

  for (int nt = 0; nt < 6; ++nt) {
    v8f acc = {};
    acc = wmma16(a0, load_b_w(nlw16, H, nt * 16, 0, lane), acc);
    acc = wmma16(a1, load_b_w(nlw16, H, nt * 16, 32, lane), acc);
    acc = wmma16(a2, load_b_w(nlw16, H, nt * 16, 64, lane), acc);
    const int j = nt * 16 + col0;
    const float b = nlb[j];
#pragma unroll
    for (int r = 0; r < 8; ++r) {
      const int i = hh * 8 + r;
      const float hv = x[(row0 + i) * H + j] + acc[r] + b;
      lds_h[i * H + j] = hv;
      rowsum[r] += hv;
      rowsq[r]  += hv * hv;
    }
  }

#pragma unroll
  for (int r = 0; r < 8; ++r) {
    float s = rowsum[r], q = rowsq[r];
#pragma unroll
    for (int m = 1; m < 16; m <<= 1) {
      s += __shfl_xor(s, m, 32);
      q += __shfl_xor(q, m, 32);
    }
    if (col0 == 0) {
      const int i = hh * 8 + r;
      const float mu = s * (1.0f / H);
      const float var = q * (1.0f / H) - mu * mu;
      lds_mu[i] = mu;
      lds_rs[i] = rsqrtf(var + 1e-5f);
    }
  }
  __syncthreads();
  for (int t = lane; t < 16 * H; t += 32) {
    const int i = t / H;
    const int j = t - i * H;
    h_out[row0 * H + t] =
        (lds_h[t] - lds_mu[i]) * lds_rs[i] * ln_g[j] + ln_b[j];
  }
}

// ---------------------------------------------------------------------------
// host launcher
// ---------------------------------------------------------------------------
extern "C" void kernel_launch(void* const* d_in, const int* in_sizes, int n_in,
                              void* d_out, int out_size, void* d_ws, size_t ws_size,
                              hipStream_t stream) {
  const float* x          = (const float*)d_in[0];
  const float* conn       = (const float*)d_in[1];
  const int*   edge_index = (const int*)d_in[2];
  const float* qkv_w      = (const float*)d_in[3];
  const float* qkv_b      = (const float*)d_in[4];
  const float* e_w        = (const float*)d_in[5];
  const float* e_b        = (const float*)d_in[6];
  const float* cl_w       = (const float*)d_in[7];
  const float* cl_b       = (const float*)d_in[8];
  const float* nl_w       = (const float*)d_in[9];
  const float* nl_b       = (const float*)d_in[10];
  const float* ln_h_g     = (const float*)d_in[11];
  const float* ln_h_b     = (const float*)d_in[12];
  const float* ln_e_g     = (const float*)d_in[13];
  const float* ln_e_b     = (const float*)d_in[14];

  float* out_h = (float*)d_out;                       // (50000, 96)
  float* out_e = out_h + (size_t)N_NODES * H;         // (800000, 96)

  // workspace layout
  char* ws = (char*)d_ws;
  float* qkv = (float*)ws;   ws += (size_t)N_NODES * 3 * H * sizeof(float);
  float* agg = (float*)ws;   ws += (size_t)N_NODES * H * sizeof(float);
  float* eagg = (float*)ws;  ws += (size_t)N_NODES * H * sizeof(float);
  _Float16* qkvw16 = (_Float16*)ws; ws += (size_t)3 * H * H * sizeof(_Float16);
  _Float16* ew16   = (_Float16*)ws; ws += (size_t)2 * H * H * sizeof(_Float16);
  _Float16* clw16  = (_Float16*)ws; ws += (size_t)H * H * sizeof(_Float16);
  _Float16* nlw16  = (_Float16*)ws; ws += (size_t)H * H * sizeof(_Float16);

  // zero the two accumulators (agg, eagg are contiguous)
  {
    const int n = 2 * N_NODES * H;
    zero_f32_kernel<<<(n + 255) / 256, 256, 0, stream>>>(agg, n);
  }
  // f32 -> f16 weight conversion (tiny)
  cvt_f16_kernel<<<(3 * H * H + 255) / 256, 256, 0, stream>>>(qkv_w, qkvw16, 3 * H * H);
  cvt_f16_kernel<<<(2 * H * H + 255) / 256, 256, 0, stream>>>(e_w, ew16, 2 * H * H);
  cvt_f16_kernel<<<(H * H + 255) / 256, 256, 0, stream>>>(cl_w, clw16, H * H);
  cvt_f16_kernel<<<(H * H + 255) / 256, 256, 0, stream>>>(nl_w, nlw16, H * H);

  qkv_gemm_kernel<<<N_NODES / 16, 32, 0, stream>>>(x, qkvw16, qkv_b, qkv);

  edge_kernel<<<N_EDGES / 16, 32, 0, stream>>>(conn, edge_index, qkv, ew16,
                                               e_b, ln_e_g, ln_e_b, eagg, agg,
                                               out_e);

  node_kernel<<<N_NODES / 16, 32, 0, stream>>>(x, agg, eagg, clw16, cl_b,
                                               nlw16, nl_b, ln_h_g, ln_h_b,
                                               out_h);
}
